// SparseAttention_40948218200590
// MI455X (gfx1250) — compile-verified, matrix operand order checked by reading, not run on hardware
//
#include <hip/hip_runtime.h>
#include <hip/hip_bf16.h>
#include <cstdint>
#include <cstddef>

// ---------------------------------------------------------------------------
// Problem constants (match reference)
//   B=128, T=256, C=384, H=6, D=64, M=128, Tt=384, WIN=64
// ---------------------------------------------------------------------------

typedef __attribute__((ext_vector_type(16))) __bf16 v16bf;
typedef __attribute__((ext_vector_type(8)))  float  v8f;

union FragBF {
    uint4 q[2];   // 32 bytes = 8 VGPRs
    v16bf v;
};

__device__ __forceinline__ unsigned short f2bf(float f) {
    unsigned u = __builtin_bit_cast(unsigned, f);
    u += 0x7FFFu + ((u >> 16) & 1u);          // round-to-nearest-even
    return (unsigned short)(u >> 16);
}

// A/B fragment for v_wmma_f32_16x16x32_bf16:
// lane L<16 : row L, K = base+0..7 (q[0]) and base+16..23 (q[1])
// lane L>=16: row L-16, K = base+8..15 and base+24..31
// caller passes p already offset by  lk = (lane>>4)*8 elements.
__device__ __forceinline__ void load_frag(FragBF& fr, const unsigned short* p) {
    fr.q[0] = *(const uint4*)(p);
    fr.q[1] = *(const uint4*)(p + 16);
}

// ---------------------------------------------------------------------------
// Prep kernels: fp32 -> bf16 conversions / layout changes
// ---------------------------------------------------------------------------

__global__ void cvt_bf16(const float* __restrict__ src, unsigned short* __restrict__ dst, int n) {
    int i = blockIdx.x * 256 + threadIdx.x;
    if (i < n) dst[i] = f2bf(src[i]);
}

// src [rows, cols] f32  ->  dst [cols, rows] bf16
__global__ void transpose_bf16(const float* __restrict__ src, unsigned short* __restrict__ dst,
                               int rows, int cols) {
    int i = blockIdx.x * 256 + threadIdx.x;
    if (i < rows * cols) {
        int r = i / cols, c = i % cols;
        dst[(size_t)c * rows + r] = f2bf(src[i]);
    }
}

// memory [128, 384] -> kmem [H=6,128,64] (row-major K rows), vmemT [H=6,64,128] (V^T rows)
__global__ void mem_prep(const float* __restrict__ mem,
                         unsigned short* __restrict__ kmem,
                         unsigned short* __restrict__ vmemT) {
    int i = blockIdx.x * 256 + threadIdx.x;
    if (i < 128 * 384) {
        int m = i / 384, c = i % 384;
        int h = c >> 6, d = c & 63;
        unsigned short bv = f2bf(mem[i]);
        kmem [((h * 128) + m) * 64 + d] = bv;
        vmemT[((h * 64) + d) * 128 + m] = bv;
    }
}

// ---------------------------------------------------------------------------
// QKV GEMM:  [32768 x 384] bf16 @ [384 x 1152] (stored transposed [1152 x 384])
// One wave per 32x64 output tile.  Epilogue scatters into attention layouts:
//   q -> qh    [B,H,T,D]  (pre-scaled by 1/sqrt(D) = 0.125)
//   k -> kself [B,H,T,D]
//   v -> vTself[B,H,D,T]  (transposed so PV B-fragments are contiguous)
// ---------------------------------------------------------------------------

__global__ __launch_bounds__(32) void qkv_gemm(
    const unsigned short* __restrict__ xbf,
    const unsigned short* __restrict__ wt,      // [1152, 384]
    const float* __restrict__ bias,             // [1152]
    unsigned short* __restrict__ qh,
    unsigned short* __restrict__ kself,
    unsigned short* __restrict__ vTself)
{
    const int lane = threadIdx.x;
    const int lm = lane & 15;
    const int lk = (lane >> 4) * 8;
    const int m0 = blockIdx.x * 32;
    const int n0 = blockIdx.y * 64;

    v8f acc[2][4] = {};

    for (int kt = 0; kt < 384; kt += 32) {
        FragBF a[2], bf[4];
        #pragma unroll
        for (int mi = 0; mi < 2; ++mi) {
            const unsigned short* p = xbf + (size_t)(m0 + mi * 16 + lm) * 384 + kt + lk;
            load_frag(a[mi], p);
            __builtin_prefetch(p + 32, 0, 1);   // next K-tile -> global_prefetch_b8
        }
        #pragma unroll
        for (int ni = 0; ni < 4; ++ni) {
            const unsigned short* p = wt + (size_t)(n0 + ni * 16 + lm) * 384 + kt + lk;
            load_frag(bf[ni], p);
        }
        #pragma unroll
        for (int mi = 0; mi < 2; ++mi)
            #pragma unroll
            for (int ni = 0; ni < 4; ++ni)
                acc[mi][ni] = __builtin_amdgcn_wmma_f32_16x16x32_bf16(
                    false, a[mi].v, false, bf[ni].v, (short)0, acc[mi][ni], false, false);
    }

    const int rowoff = (lane >> 4) * 8;
    #pragma unroll
    for (int ni = 0; ni < 4; ++ni) {
        const int col = n0 + ni * 16 + lm;
        const float bv = bias[col];
        const int which = col / 384;            // 0:q 1:k 2:v (64-col tile never straddles)
        const int c = col % 384;
        const int h = c >> 6, d = c & 63;
        #pragma unroll
        for (int mi = 0; mi < 2; ++mi) {
            #pragma unroll
            for (int r = 0; r < 8; ++r) {
                const int row = m0 + mi * 16 + rowoff + r;   // global (b*T + t)
                const int bb = row >> 8;
                const int t  = row & 255;
                const size_t bh = (size_t)(bb * 6 + h);
                const float val = acc[mi][ni][r] + bv;
                if (which == 0)
                    qh   [(bh * 256 + t) * 64 + d] = f2bf(val * 0.125f);
                else if (which == 1)
                    kself[(bh * 256 + t) * 64 + d] = f2bf(val);
                else
                    vTself[(bh * 64 + d) * 256 + t] = f2bf(val);
            }
        }
    }
}

// ---------------------------------------------------------------------------
// Attention: one wave per (b, h, 16-row query block).
//  S = Q K^T (WMMA, K rows ARE the B-fragments since B = K^T) -> LDS f32
//  masked softmax in LDS (memory prefix unmasked; self: causal AND |i-j|<=64)
//  O = P V (A-fragments from padded LDS bf16 P; B-fragments from V^T rows)
// ---------------------------------------------------------------------------

__global__ __launch_bounds__(32) void attn_kernel(
    const unsigned short* __restrict__ qh,
    const unsigned short* __restrict__ kself,
    const unsigned short* __restrict__ vTself,
    const unsigned short* __restrict__ kmem,
    const unsigned short* __restrict__ vmemT,
    const float* __restrict__ gate,
    unsigned short* __restrict__ ygated)
{
    __shared__ __align__(16) float          S[16][388];   // +4 pad
    __shared__ __align__(16) unsigned short P[16][392];   // pitch 392 -> conflict-free b128
    __shared__ float rinv[16];

    const int lane = threadIdx.x;
    const int lm = lane & 15;
    const int lk = (lane >> 4) * 8;
    const int i0 = blockIdx.x * 16;   // query block start
    const int h  = blockIdx.y;
    const int b  = blockIdx.z;
    const size_t bh = (size_t)(b * 6 + h);

    // Q fragments: rows i0..i0+15, two K-tiles covering D=64 (scale pre-folded)
    FragBF qf[2];
    {
        const unsigned short* qrow = qh + (bh * 256 + (i0 + lm)) * 64;
        load_frag(qf[0], qrow + 0  + lk);
        load_frag(qf[1], qrow + 32 + lk);
    }

    // ---- Phase 1: S tiles (24 key tiles of 16; tiles 0..7 = memory prefix) ----
    for (int nt = 0; nt < 24; ++nt) {
        const unsigned short* krow = (nt < 8)
            ? kmem  + ((size_t)h * 128 + (nt * 16 + lm)) * 64
            : kself + (bh * 256 + (nt * 16 - 128 + lm)) * 64;
        FragBF kf0, kf1;
        load_frag(kf0, krow + 0  + lk);
        load_frag(kf1, krow + 32 + lk);
        v8f acc = {};
        acc = __builtin_amdgcn_wmma_f32_16x16x32_bf16(false, qf[0].v, false, kf0.v, (short)0, acc, false, false);
        acc = __builtin_amdgcn_wmma_f32_16x16x32_bf16(false, qf[1].v, false, kf1.v, (short)0, acc, false, false);
        const int rowbase = (lane >> 4) * 8;
        #pragma unroll
        for (int r = 0; r < 8; ++r)
            S[rowbase + r][nt * 16 + lm] = acc[r];
    }
    __syncthreads();

    // ---- Phase 2: masked softmax, 2 lanes per row ----
    {
        const int row  = lane >> 1;
        const int half = lane & 1;
        const int j0 = half * 192, j1 = j0 + 192;
        const int i = i0 + row;
        float mx = -3.0e38f;
        for (int j = j0; j < j1; ++j) {
            float s = S[row][j];
            if (j >= 128) {
                const int js = j - 128;
                if (js > i || (i - js) > 64) s = -3.0e38f;
            }
            S[row][j] = s;
            mx = fmaxf(mx, s);
        }
        mx = fmaxf(mx, __shfl_xor(mx, 1, 32));
        float sum = 0.0f;
        for (int j = j0; j < j1; ++j) {
            const float e = __expf(S[row][j] - mx);
            sum += e;
            P[row][j] = f2bf(e);
        }
        sum += __shfl_xor(sum, 1, 32);
        if (half == 0) rinv[row] = 1.0f / sum;
    }
    __syncthreads();

    // ---- Phase 3: O = P V over 12 K-tiles of 32 (tiles 0..3 = memory prefix) ----
    v8f o[4] = {};
    for (int kt = 0; kt < 12; ++kt) {
        FragBF pa;
        const unsigned short* pp = &P[lm][kt * 32 + lk];
        pa.q[0] = *(const uint4*)(pp);
        pa.q[1] = *(const uint4*)(pp + 16);
        #pragma unroll
        for (int ni = 0; ni < 4; ++ni) {
            const int d = ni * 16 + lm;
            const unsigned short* vp = (kt < 4)
                ? vmemT  + ((size_t)h * 64 + d) * 128 + kt * 32 + lk
                : vTself + (bh * 64 + d) * 256 + (kt - 4) * 32 + lk;
            FragBF vf;
            load_frag(vf, vp);
            o[ni] = __builtin_amdgcn_wmma_f32_16x16x32_bf16(
                false, pa.v, false, vf.v, (short)0, o[ni], false, false);
        }
    }

    // ---- Epilogue: 1/rowsum, gate, bf16 store to [B*T, C] ----
    const int rowbase = (lane >> 4) * 8;
    #pragma unroll
    for (int ni = 0; ni < 4; ++ni) {
        const int d = ni * 16 + lm;
        const int c = h * 64 + d;
        const float g = gate[c];
        #pragma unroll
        for (int r = 0; r < 8; ++r) {
            const int row = rowbase + r;
            const float val = o[ni][r] * rinv[row] * g;
            ygated[((size_t)b * 256 + i0 + row) * 384 + c] = f2bf(val);
        }
    }
}

// ---------------------------------------------------------------------------
// Output projection: [32768 x 384] bf16 @ [384 x 384] (stored transposed) + bias
// ---------------------------------------------------------------------------

__global__ __launch_bounds__(32) void proj_gemm(
    const unsigned short* __restrict__ yg,
    const unsigned short* __restrict__ wt,      // [384, 384] (W_proj^T)
    const float* __restrict__ bias,             // [384]
    float* __restrict__ out)
{
    const int lane = threadIdx.x;
    const int lm = lane & 15;
    const int lk = (lane >> 4) * 8;
    const int m0 = blockIdx.x * 32;
    const int n0 = blockIdx.y * 64;

    v8f acc[2][4] = {};

    for (int kt = 0; kt < 384; kt += 32) {
        FragBF a[2], bf[4];
        #pragma unroll
        for (int mi = 0; mi < 2; ++mi) {
            const unsigned short* p = yg + (size_t)(m0 + mi * 16 + lm) * 384 + kt + lk;
            load_frag(a[mi], p);
        }
        #pragma unroll
        for (int ni = 0; ni < 4; ++ni) {
            const unsigned short* p = wt + (size_t)(n0 + ni * 16 + lm) * 384 + kt + lk;
            load_frag(bf[ni], p);
        }
        #pragma unroll
        for (int mi = 0; mi < 2; ++mi)
            #pragma unroll
            for (int ni = 0; ni < 4; ++ni)
                acc[mi][ni] = __builtin_amdgcn_wmma_f32_16x16x32_bf16(
                    false, a[mi].v, false, bf[ni].v, (short)0, acc[mi][ni], false, false);
    }

    const int rowoff = (lane >> 4) * 8;
    #pragma unroll
    for (int ni = 0; ni < 4; ++ni) {
        const int col = n0 + ni * 16 + lm;
        const float bv = bias[col];
        #pragma unroll
        for (int mi = 0; mi < 2; ++mi)
            #pragma unroll
            for (int r = 0; r < 8; ++r)
                out[(size_t)(m0 + mi * 16 + rowoff + r) * 384 + col] = acc[mi][ni][r] + bv;
    }
}

// ---------------------------------------------------------------------------
// Launch
// ---------------------------------------------------------------------------

extern "C" void kernel_launch(void* const* d_in, const int* in_sizes, int n_in,
                              void* d_out, int out_size, void* d_ws, size_t ws_size,
                              hipStream_t stream)
{
    (void)in_sizes; (void)n_in; (void)out_size; (void)ws_size;

    const float* x     = (const float*)d_in[0];   // [128,256,384]
    const float* mem   = (const float*)d_in[1];   // [128,384]
    const float* Wqkv  = (const float*)d_in[2];   // [384,1152]
    const float* bqkv  = (const float*)d_in[3];   // [1152]
    const float* Wproj = (const float*)d_in[4];   // [384,384]
    const float* bproj = (const float*)d_in[5];   // [384]
    const float* gate  = (const float*)d_in[6];   // [384]
    float* out = (float*)d_out;                   // [128,256,384] f32

    char* w = (char*)d_ws;
    size_t off = 0;
    auto take = [&](size_t bytes) -> void* {
        void* p = w + off;
        off += (bytes + 255) & ~(size_t)255;
        return p;
    };

    unsigned short* xbf    = (unsigned short*)take(32768ull * 384 * 2);
    unsigned short* wqkvt  = (unsigned short*)take(1152ull  * 384 * 2);
    unsigned short* wprojt = (unsigned short*)take(384ull   * 384 * 2);
    unsigned short* kmem   = (unsigned short*)take(6ull * 128 * 64 * 2);
    unsigned short* vmemT  = (unsigned short*)take(6ull * 64 * 128 * 2);
    unsigned short* qh     = (unsigned short*)take(128ull * 6 * 256 * 64 * 2);
    unsigned short* kself  = (unsigned short*)take(128ull * 6 * 256 * 64 * 2);
    unsigned short* vTself = (unsigned short*)take(128ull * 6 * 64 * 256 * 2);
    unsigned short* yg     = (unsigned short*)take(32768ull * 384 * 2);

    cvt_bf16<<<(12582912 + 255) / 256, 256, 0, stream>>>(x, xbf, 12582912);
    transpose_bf16<<<(442368 + 255) / 256, 256, 0, stream>>>(Wqkv, wqkvt, 384, 1152);
    transpose_bf16<<<(147456 + 255) / 256, 256, 0, stream>>>(Wproj, wprojt, 384, 384);
    mem_prep<<<(49152 + 255) / 256, 256, 0, stream>>>(mem, kmem, vmemT);

    qkv_gemm<<<dim3(1024, 18), 32, 0, stream>>>(xbf, wqkvt, bqkv, qh, kself, vTself);
    attn_kernel<<<dim3(16, 6, 128), 32, 0, stream>>>(qh, kself, vTself, kmem, vmemT, gate, yg);
    proj_gemm<<<dim3(1024, 6), 32, 0, stream>>>(yg, wprojt, bproj, out);
}
